// HierarchicalAttentionBlock_57681410785556
// MI455X (gfx1250) — matrix operand order repaired and checked by reading, not compiled, not run on hardware
//
#include <hip/hip_runtime.h>
#include <hip/hip_bf16.h>

#define B_  4
#define S_  1024
#define H_  2048
#define NH_ 16
#define HD_ 128
#define PS_ 16
#define P_  64
#define T_  (B_ * S_)
#define EPS_ 1e-5f

typedef unsigned short u16;
typedef __attribute__((ext_vector_type(16))) __bf16 v16bf;
typedef __attribute__((ext_vector_type(8)))  float  v8f;

struct alignas(16) U128 { unsigned int x, y, z, w; };
union FragBF { v16bf v; U128 q[2]; u16 e[16]; };

__device__ __forceinline__ u16 f2bf(float f) {
  unsigned int u = __float_as_uint(f);
  u += 0x7fffu + ((u >> 16) & 1u);          // round-to-nearest-even
  return (u16)(u >> 16);
}
__device__ __forceinline__ float bf2f(u16 h) {
  return __uint_as_float(((unsigned int)h) << 16);
}

__device__ __forceinline__ v8f wmma_bf16(v16bf a, v16bf b, v8f c) {
  return __builtin_amdgcn_wmma_f32_16x16x32_bf16(false, a, false, b, (short)0, c,
                                                 false, false);
}

// A fragment: 16x32 bf16, lane L: row M=L&15; lanes<16 hold K 0..7 & 16..23,
// lanes>=16 hold K 8..15 & 24..31  (ISA 7.12.2 16-bit A layout)
__device__ __forceinline__ v16bf load_a_frag(const u16* A, int lda, int row0,
                                             int k, int lane) {
  int m  = row0 + (lane & 15);
  int kb = (lane < 16) ? 0 : 8;
  const u16* p = A + (size_t)m * lda + k + kb;
  FragBF f;
  f.q[0] = *(const U128*)(p);
  f.q[1] = *(const U128*)(p + 16);
  return f.v;
}
// B fragment: 32x16 bf16, lane L: col N=L&15; lanes<16 hold K 0..7 & 16..23,
// lanes>=16 hold K 8..15 & 24..31 (matches A fragment layout)
__device__ __forceinline__ v16bf load_b_frag(const u16* Bw, int ldb, int col0,
                                             int k, int lane) {
  int n  = col0 + (lane & 15);
  int kb = (lane < 16) ? 0 : 8;
  const u16* p = Bw + (size_t)n * ldb + k + kb;
  FragBF f;
  f.q[0] = *(const U128*)(p);
  f.q[1] = *(const U128*)(p + 16);
  return f.v;
}
// A fragment sourced from f32 LDS (softmax probabilities), converted on the fly
__device__ __forceinline__ v16bf load_a_frag_lds(const float* sm, int ldm,
                                                 int row0, int k, int lane) {
  int m  = row0 + (lane & 15);
  int kb = (lane < 16) ? 0 : 8;
  const float* p = sm + (size_t)m * ldm + k + kb;
  FragBF f;
#pragma unroll
  for (int j = 0; j < 8; ++j) { f.e[j] = f2bf(p[j]); f.e[8 + j] = f2bf(p[16 + j]); }
  return f.v;
}

// 32x32 output tile per wave, K-major A and B, f32 accumulation
__device__ __forceinline__ void core32x32(const u16* A, int lda, const u16* Bw,
                                          int ldb, int row0, int col0, int K,
                                          int lane, v8f acc[2][2]) {
  for (int k = 0; k < K; k += 32) {
    if (k + 32 < K) {
      __builtin_prefetch(A + (size_t)(row0 + (lane & 15)) * lda + k + 32, 0, 1);
      __builtin_prefetch(Bw + (size_t)(col0 + (lane & 15)) * ldb + k + 32, 0, 1);
    }
    v16bf a0 = load_a_frag(A, lda, row0, k, lane);
    v16bf a1 = load_a_frag(A, lda, row0 + 16, k, lane);
    v16bf b0 = load_b_frag(Bw, ldb, col0, k, lane);
    v16bf b1 = load_b_frag(Bw, ldb, col0 + 16, k, lane);
    acc[0][0] = wmma_bf16(a0, b0, acc[0][0]);
    acc[0][1] = wmma_bf16(a0, b1, acc[0][1]);
    acc[1][0] = wmma_bf16(a1, b0, acc[1][0]);
    acc[1][1] = wmma_bf16(a1, b1, acc[1][1]);
  }
}

__global__ void cvt_bf16k(const float* __restrict__ src, u16* __restrict__ dst,
                          size_t n) {
  size_t i = (size_t)blockIdx.x * blockDim.x + threadIdx.x;
  size_t stride = (size_t)gridDim.x * blockDim.x;
  for (; i < n; i += stride) dst[i] = f2bf(src[i]);
}

// para_states = mean over each PS-token block, output bf16 [B,P,H]
__global__ void pool_kernel(const float* __restrict__ x, u16* __restrict__ poolb) {
  int i = blockIdx.x * blockDim.x + threadIdx.x;
  if (i >= B_ * P_ * H_) return;
  int hh = i % H_;
  int p  = (i / H_) % P_;
  int b  = i / (H_ * P_);
  const float* base = x + ((size_t)b * S_ + p * PS_) * H_ + hh;
  float s = 0.f;
#pragma unroll
  for (int j = 0; j < PS_; ++j) s += base[(size_t)j * H_];
  poolb[i] = f2bf(s * (1.f / PS_));
}

// QKV projection: out cols [0,H)=Q, [H,2H)=K, [2H,3H)=V(transposed store)
__global__ void gemm_qkv(const u16* __restrict__ A, const u16* __restrict__ Bw,
                         const float* __restrict__ bias, u16* __restrict__ qd,
                         u16* __restrict__ kd, u16* __restrict__ vTd, int Sq,
                         int K) {
  int lane = threadIdx.x & 31, wave = threadIdx.x >> 5;
  int row0 = blockIdx.y * 64 + (wave >> 1) * 32;
  int col0 = blockIdx.x * 64 + (wave & 1) * 32;
  v8f acc[2][2] = {};
  core32x32(A, K, Bw, K, row0, col0, K, lane, acc);
  int mofs = (lane < 16) ? 0 : 8;
  int nl = lane & 15;
#pragma unroll
  for (int ai = 0; ai < 2; ++ai)
#pragma unroll
    for (int bi = 0; bi < 2; ++bi) {
      int n = col0 + bi * 16 + nl;
      int which = n / H_;
      int rem = n - which * H_;
      int hh = rem / HD_, d = rem - (rem / HD_) * HD_;
      float bv = bias[n];
#pragma unroll
      for (int r = 0; r < 8; ++r) {
        int m = row0 + ai * 16 + mofs + r;
        int bb = m / Sq, s = m - bb * Sq;
        u16 hv = f2bf(acc[ai][bi][r] + bv);
        if (which == 2)
          vTd[((size_t)(bb * NH_ + hh) * HD_ + d) * Sq + s] = hv;
        else if (which == 1)
          kd[((size_t)(bb * NH_ + hh) * Sq + s) * HD_ + d] = hv;
        else
          qd[((size_t)(bb * NH_ + hh) * Sq + s) * HD_ + d] = hv;
      }
    }
}

// Fused attention: one block = 64 query rows of one (b,h).
// Score panel 64 x Sk lives in dynamic LDS (f32), softmax in place, then P@V.
__global__ void flash_attn(const u16* __restrict__ q, const u16* __restrict__ kk,
                           const u16* __restrict__ vT, u16* __restrict__ ao,
                           int Sq, int Sk) {
  extern __shared__ float sm[];  // 64 * Sk floats
  int bh = blockIdx.y;
  int bb = bh / NH_, hh = bh % NH_;
  int s0 = blockIdx.x * 64;
  const u16* Q  = q  + (size_t)bh * Sq * HD_ + (size_t)s0 * HD_;
  const u16* Km = kk + (size_t)bh * Sk * HD_;
  const u16* V  = vT + (size_t)bh * HD_ * Sk;
  int lane = threadIdx.x & 31, wave = threadIdx.x >> 5;  // 8 waves

  // Phase 1: scores = Q K^T  (64 x Sk), tiles round-robin over waves
  int ntiles = (Sk >> 5) * 2;
  for (int t = wave; t < ntiles; t += 8) {
    int row0 = (t & 1) * 32;
    int col0 = (t >> 1) * 32;
    v8f acc[2][2] = {};
    core32x32(Q, HD_, Km, HD_, row0, col0, HD_, lane, acc);
    int mofs = (lane < 16) ? 0 : 8;
    int nl = lane & 15;
#pragma unroll
    for (int ai = 0; ai < 2; ++ai)
#pragma unroll
      for (int bi = 0; bi < 2; ++bi)
#pragma unroll
        for (int r = 0; r < 8; ++r)
          sm[(size_t)(row0 + ai * 16 + mofs + r) * Sk + col0 + bi * 16 + nl] =
              acc[ai][bi][r];
  }
  __syncthreads();

  // Phase 2: scaled softmax per row (wave handles 8 rows, lane-parallel)
  const float scale = 0.08838834764831845f;  // 1/sqrt(128)
  for (int r = wave * 8; r < wave * 8 + 8; ++r) {
    float* row = sm + (size_t)r * Sk;
    float mx = -3.0e38f;
    for (int i = lane; i < Sk; i += 32) mx = fmaxf(mx, row[i]);
    for (int o = 16; o; o >>= 1) mx = fmaxf(mx, __shfl_xor(mx, o, 32));
    float sum = 0.f;
    for (int i = lane; i < Sk; i += 32) {
      float e = __expf((row[i] - mx) * scale);
      row[i] = e;
      sum += e;
    }
    for (int o = 16; o; o >>= 1) sum += __shfl_xor(sum, o, 32);
    float inv = 1.f / sum;
    for (int i = lane; i < Sk; i += 32) row[i] *= inv;
  }
  __syncthreads();

  // Phase 3: O = P V  (64 x 128); each wave one 32x32 tile
  {
    int row0 = (wave >> 2) * 32;
    int col0 = (wave & 3) * 32;
    v8f acc[2][2] = {};
    for (int k = 0; k < Sk; k += 32) {
      v16bf a0 = load_a_frag_lds(sm, Sk, row0, k, lane);
      v16bf a1 = load_a_frag_lds(sm, Sk, row0 + 16, k, lane);
      v16bf b0 = load_b_frag(V, Sk, col0, k, lane);
      v16bf b1 = load_b_frag(V, Sk, col0 + 16, k, lane);
      acc[0][0] = wmma_bf16(a0, b0, acc[0][0]);
      acc[0][1] = wmma_bf16(a0, b1, acc[0][1]);
      acc[1][0] = wmma_bf16(a1, b0, acc[1][0]);
      acc[1][1] = wmma_bf16(a1, b1, acc[1][1]);
    }
    int mofs = (lane < 16) ? 0 : 8, nl = lane & 15;
#pragma unroll
    for (int ai = 0; ai < 2; ++ai)
#pragma unroll
      for (int bi = 0; bi < 2; ++bi)
#pragma unroll
        for (int r = 0; r < 8; ++r) {
          int s = s0 + row0 + ai * 16 + mofs + r;
          int d = col0 + bi * 16 + nl;
          ao[((size_t)(bb * Sq + s)) * H_ + hh * HD_ + d] = f2bf(acc[ai][bi][r]);
        }
  }
}

// NT GEMM with epilogue: MODE 0 = bias->bf16, 1 = bias->GELU->bf16, 2 = bias->f32
template <int MODE>
__global__ void gemm_nt(const u16* __restrict__ A, const u16* __restrict__ Bw,
                        const float* __restrict__ bias, void* __restrict__ Cout,
                        int M, int N, int K) {
  int lane = threadIdx.x & 31, wave = threadIdx.x >> 5;
  int row0 = blockIdx.y * 64 + (wave >> 1) * 32;
  int col0 = blockIdx.x * 64 + (wave & 1) * 32;
  v8f acc[2][2] = {};
  core32x32(A, K, Bw, K, row0, col0, K, lane, acc);
  int mofs = (lane < 16) ? 0 : 8;
  int nl = lane & 15;
#pragma unroll
  for (int ai = 0; ai < 2; ++ai)
#pragma unroll
    for (int bi = 0; bi < 2; ++bi) {
      int n = col0 + bi * 16 + nl;
      float bv = bias[n];
#pragma unroll
      for (int r = 0; r < 8; ++r) {
        int m = row0 + ai * 16 + mofs + r;
        float v = acc[ai][bi][r] + bv;
        if (MODE == 1) v = 0.5f * v * (1.0f + erff(v * 0.70710678f));
        size_t idx = (size_t)m * N + n;
        if (MODE == 2) ((float*)Cout)[idx] = v;
        else           ((u16*)Cout)[idx] = f2bf(v);
      }
    }
}

// fin = [doc | para_tok | sent]  (bf16 copies, para expanded by token/PS)
__global__ void build_fin(const u16* __restrict__ docb, const u16* __restrict__ parab,
                          const u16* __restrict__ sentb, u16* __restrict__ fin) {
  size_t i = (size_t)blockIdx.x * blockDim.x + threadIdx.x;
  if (i >= (size_t)T_ * 3 * H_) return;
  int c = (int)(i % (3 * H_));
  int t = (int)(i / (3 * H_));
  int bb = t / S_, s = t - bb * S_;
  u16 v;
  if (c < H_)          v = docb[(size_t)t * H_ + c];
  else if (c < 2 * H_) v = parab[((size_t)bb * P_ + s / PS_) * H_ + (c - H_)];
  else                 v = sentb[(size_t)t * H_ + (c - 2 * H_)];
  fin[i] = v;
}

// per-token: logits = g . w2^T + b2, softmax3, fused = w0*doc + w1*para + w2*sent
__global__ void gate_fuse(const u16* __restrict__ g, const float* __restrict__ w2,
                          const float* __restrict__ b2, const u16* __restrict__ docb,
                          const u16* __restrict__ parab, const u16* __restrict__ sentb,
                          u16* __restrict__ fused) {
  int t = blockIdx.x, tid = threadIdx.x;
  int bb = t / S_, s = t - bb * S_;
  size_t prow = ((size_t)bb * P_ + s / PS_) * H_;
  __shared__ float red[3][256];
  float p0 = 0.f, p1 = 0.f, p2 = 0.f;
  const u16* gr = g + (size_t)t * (2 * H_);
  for (int k = tid; k < 2 * H_; k += 256) {
    float gv = bf2f(gr[k]);
    p0 += gv * w2[k];
    p1 += gv * w2[2 * H_ + k];
    p2 += gv * w2[4 * H_ + k];
  }
  red[0][tid] = p0; red[1][tid] = p1; red[2][tid] = p2;
  __syncthreads();
  for (int st = 128; st; st >>= 1) {
    if (tid < st) {
      red[0][tid] += red[0][tid + st];
      red[1][tid] += red[1][tid + st];
      red[2][tid] += red[2][tid + st];
    }
    __syncthreads();
  }
  __shared__ float wsh[3];
  if (tid == 0) {
    float l0 = red[0][0] + b2[0], l1 = red[1][0] + b2[1], l2 = red[2][0] + b2[2];
    float mx = fmaxf(l0, fmaxf(l1, l2));
    float e0 = __expf(l0 - mx), e1 = __expf(l1 - mx), e2 = __expf(l2 - mx);
    float inv = 1.f / (e0 + e1 + e2);
    wsh[0] = e0 * inv; wsh[1] = e1 * inv; wsh[2] = e2 * inv;
  }
  __syncthreads();
  float w0 = wsh[0], w1 = wsh[1], wv2 = wsh[2];
  for (int hc = tid; hc < H_; hc += 256) {
    float v = w0 * bf2f(docb[(size_t)t * H_ + hc]) +
              w1 * bf2f(parab[prow + hc]) +
              wv2 * bf2f(sentb[(size_t)t * H_ + hc]);
    fused[(size_t)t * H_ + hc] = f2bf(v);
  }
}

// out = LayerNorm(x + proj) * gamma + beta
__global__ void ln_residual(const float* __restrict__ x, const float* __restrict__ proj,
                            const float* __restrict__ gw, const float* __restrict__ bw,
                            float* __restrict__ out) {
  int t = blockIdx.x, tid = threadIdx.x;
  __shared__ float r1[256], r2[256];
  const float* xr = x + (size_t)t * H_;
  const float* pr = proj + (size_t)t * H_;
  float s = 0.f, ss = 0.f;
  for (int i = tid; i < H_; i += 256) {
    float y = xr[i] + pr[i];
    s += y; ss += y * y;
  }
  r1[tid] = s; r2[tid] = ss;
  __syncthreads();
  for (int st = 128; st; st >>= 1) {
    if (tid < st) { r1[tid] += r1[tid + st]; r2[tid] += r2[tid + st]; }
    __syncthreads();
  }
  float mean = r1[0] * (1.f / H_);
  float var = r2[0] * (1.f / H_) - mean * mean;
  float rstd = rsqrtf(var + EPS_);
  for (int i = tid; i < H_; i += 256) {
    float y = xr[i] + pr[i];
    out[(size_t)t * H_ + i] = (y - mean) * rstd * gw[i] + bw[i];
  }
}

extern "C" void kernel_launch(void* const* d_in, const int* in_sizes, int n_in,
                              void* d_out, int out_size, void* d_ws, size_t ws_size,
                              hipStream_t stream) {
  (void)in_sizes; (void)n_in; (void)out_size; (void)ws_size;
  const float* x          = (const float*)d_in[0];
  const float* doc_w_in   = (const float*)d_in[1];
  const float* doc_b_in   = (const float*)d_in[2];
  const float* doc_w_out  = (const float*)d_in[3];
  const float* doc_b_out  = (const float*)d_in[4];
  const float* para_w_in  = (const float*)d_in[5];
  const float* para_b_in  = (const float*)d_in[6];
  const float* para_w_out = (const float*)d_in[7];
  const float* para_b_out = (const float*)d_in[8];
  const float* sent_w_in  = (const float*)d_in[9];
  const float* sent_b_in  = (const float*)d_in[10];
  const float* sent_w_out = (const float*)d_in[11];
  const float* sent_b_out = (const float*)d_in[12];
  const float* fuse_w1    = (const float*)d_in[13];
  const float* fuse_b1    = (const float*)d_in[14];
  const float* fuse_w2    = (const float*)d_in[15];
  const float* fuse_b2    = (const float*)d_in[16];
  const float* out_w      = (const float*)d_in[17];
  const float* out_b      = (const float*)d_in[18];
  const float* ln_g       = (const float*)d_in[19];
  const float* ln_b       = (const float*)d_in[20];

  size_t off = 0;
  auto alloc = [&](size_t elems, size_t esz) -> void* {
    void* p = (void*)((char*)d_ws + off);
    off += (elems * esz + 255) & ~(size_t)255;
    return p;
  };
  u16* xb        = (u16*)alloc((size_t)T_ * H_, 2);
  u16* wdin_b    = (u16*)alloc((size_t)3 * H_ * H_, 2);
  u16* wpin_b    = (u16*)alloc((size_t)3 * H_ * H_, 2);
  u16* wsin_b    = (u16*)alloc((size_t)3 * H_ * H_, 2);
  u16* wdout_b   = (u16*)alloc((size_t)H_ * H_, 2);
  u16* wpout_b   = (u16*)alloc((size_t)H_ * H_, 2);
  u16* wsout_b   = (u16*)alloc((size_t)H_ * H_, 2);
  u16* fuse1_b   = (u16*)alloc((size_t)2 * H_ * 3 * H_, 2);
  u16* outw_b    = (u16*)alloc((size_t)H_ * H_, 2);
  u16* poolb     = (u16*)alloc((size_t)B_ * P_ * H_, 2);
  u16* qb        = (u16*)alloc((size_t)T_ * H_, 2);
  u16* kbuf      = (u16*)alloc((size_t)T_ * H_, 2);
  u16* vtb       = (u16*)alloc((size_t)T_ * H_, 2);
  u16* aob       = (u16*)alloc((size_t)T_ * H_, 2);
  u16* docb      = (u16*)alloc((size_t)T_ * H_, 2);
  u16* parab     = (u16*)alloc((size_t)B_ * P_ * H_, 2);
  u16* sentb     = (u16*)alloc((size_t)T_ * H_, 2);
  u16* finb      = (u16*)alloc((size_t)T_ * 3 * H_, 2);
  u16* gbuf      = (u16*)alloc((size_t)T_ * 2 * H_, 2);
  u16* fusedb    = (u16*)alloc((size_t)T_ * H_, 2);
  float* projb   = (float*)alloc((size_t)T_ * H_, 4);

  auto cvt = [&](const float* s, u16* dp, size_t n) {
    int blocks = (int)((n + 255) / 256);
    cvt_bf16k<<<blocks, 256, 0, stream>>>(s, dp, n);
  };
  cvt(x, xb, (size_t)T_ * H_);
  cvt(doc_w_in,  wdin_b,  (size_t)3 * H_ * H_);
  cvt(para_w_in, wpin_b,  (size_t)3 * H_ * H_);
  cvt(sent_w_in, wsin_b,  (size_t)3 * H_ * H_);
  cvt(doc_w_out,  wdout_b, (size_t)H_ * H_);
  cvt(para_w_out, wpout_b, (size_t)H_ * H_);
  cvt(sent_w_out, wsout_b, (size_t)H_ * H_);
  cvt(fuse_w1, fuse1_b, (size_t)2 * H_ * 3 * H_);
  cvt(out_w, outw_b, (size_t)H_ * H_);

  // allow 256KB dynamic LDS for the full-sequence attention blocks
  (void)hipFuncSetAttribute((const void*)flash_attn,
                            hipFuncAttributeMaxDynamicSharedMemorySize,
                            64 * S_ * 4);

  // ---- document-level attention ----
  gemm_qkv<<<dim3(3 * H_ / 64, T_ / 64), 128, 0, stream>>>(
      xb, wdin_b, doc_b_in, qb, kbuf, vtb, S_, H_);
  flash_attn<<<dim3(S_ / 64, B_ * NH_), 256, 64 * S_ * 4, stream>>>(
      qb, kbuf, vtb, aob, S_, S_);
  gemm_nt<0><<<dim3(H_ / 64, T_ / 64), 128, 0, stream>>>(
      aob, wdout_b, doc_b_out, (void*)docb, T_, H_, H_);

  // ---- paragraph-level attention ----
  {
    int n = B_ * P_ * H_;
    pool_kernel<<<(n + 255) / 256, 256, 0, stream>>>(x, poolb);
  }
  gemm_qkv<<<dim3(3 * H_ / 64, (B_ * P_) / 64), 128, 0, stream>>>(
      poolb, wpin_b, para_b_in, qb, kbuf, vtb, P_, H_);
  flash_attn<<<dim3(1, B_ * NH_), 256, 64 * P_ * 4, stream>>>(
      qb, kbuf, vtb, aob, P_, P_);
  gemm_nt<0><<<dim3(H_ / 64, (B_ * P_) / 64), 128, 0, stream>>>(
      aob, wpout_b, para_b_out, (void*)parab, B_ * P_, H_, H_);

  // ---- sentence-level attention ----
  gemm_qkv<<<dim3(3 * H_ / 64, T_ / 64), 128, 0, stream>>>(
      xb, wsin_b, sent_b_in, qb, kbuf, vtb, S_, H_);
  flash_attn<<<dim3(S_ / 64, B_ * NH_), 256, 64 * S_ * 4, stream>>>(
      qb, kbuf, vtb, aob, S_, S_);
  gemm_nt<0><<<dim3(H_ / 64, T_ / 64), 128, 0, stream>>>(
      aob, wsout_b, sent_b_out, (void*)sentb, T_, H_, H_);

  // ---- gated fusion ----
  {
    size_t n = (size_t)T_ * 3 * H_;
    build_fin<<<(int)((n + 255) / 256), 256, 0, stream>>>(docb, parab, sentb, finb);
  }
  gemm_nt<1><<<dim3(2 * H_ / 64, T_ / 64), 128, 0, stream>>>(
      finb, fuse1_b, fuse_b1, (void*)gbuf, T_, 2 * H_, 3 * H_);
  gate_fuse<<<T_, 256, 0, stream>>>(gbuf, fuse_w2, fuse_b2, docb, parab, sentb,
                                    fusedb);
  gemm_nt<2><<<dim3(H_ / 64, T_ / 64), 128, 0, stream>>>(
      fusedb, outw_b, out_b, (void*)projb, T_, H_, H_);
  ln_residual<<<T_, 256, 0, stream>>>(x, projb, ln_g, ln_b, (float*)d_out);
}
